// Transformer_47193100648577
// MI455X (gfx1250) — matrix operand order repaired and checked by reading, not compiled, not run on hardware
//
#include <hip/hip_runtime.h>
#include <hip/hip_bf16.h>
#include <math.h>

// ---------------------------------------------------------------------------
// Model constants (match reference)
// ---------------------------------------------------------------------------
#define DMODEL 512
#define NHEAD  8
#define DKH    64      // DMODEL / NHEAD
#define NLAYER 6
#define FDIM   2048
#define BSZ    4
#define SEQ    512     // S == T == 512
#define NTOK   (BSZ * SEQ)   // 2048 tokens per stream
#define VOCAB  32000

typedef __attribute__((ext_vector_type(16))) _Float16 v16h;
typedef __attribute__((ext_vector_type(8)))  _Float16 v8h;
typedef __attribute__((ext_vector_type(8)))  float    v8f;

// ---------------------------------------------------------------------------
// Embedding + positional encoding:  out[t, d] = emb[tok[t], d]*sqrt(D) + pe[s, d]
// ---------------------------------------------------------------------------
__global__ __launch_bounds__(256) void embed_kernel(
    const int* __restrict__ tok, const float* __restrict__ emb,
    const float* __restrict__ pe, float* __restrict__ out)
{
  const int t   = blockIdx.x;          // token index in [0, NTOK)
  const int s   = t % SEQ;             // position within sequence
  const int tid = threadIdx.x;
  const int id  = tok[t];
  const float scale = 22.627416997969522f;  // sqrt(512)
  const size_t ob = (size_t)t * DMODEL;
  const size_t eb = (size_t)id * DMODEL;
  const size_t pb = (size_t)s * DMODEL;
  out[ob + tid]        = emb[eb + tid]        * scale + pe[pb + tid];
  out[ob + tid + 256]  = emb[eb + tid + 256]  * scale + pe[pb + tid + 256];
}

// ---------------------------------------------------------------------------
// WMMA GEMM:  C[M,N] = act(A[M,K] @ W[K,N] + bias[N])
//   Block tile 128(M) x 64(N), 8 waves in 4x2 grid, each wave owns 32x32
//   (2 A-frags x 2 B-frags -> 4 wmmas per K-step).  K step 32.
//   LDS layouts are fragment-ready so every fragment is two ds_load_b128:
//     As [128][40]  row-major  (row stride 80B, 16B aligned)
//     Bts[ 64][40]  TRANSPOSED ([n][k]); staged via coalesced column loads
//                   (8 x global_load_b32 per thread, each instruction reads a
//                   full 128B row segment across the wave) + one ds_store_b128.
//   Requires M%128==0, N%64==0, K%32==0 (true everywhere here).
//   act: 0 = none, 1 = exact GELU.
// ---------------------------------------------------------------------------
#define BM 128
#define BN 64
#define BK 32
#define LDK (BK + 8)   // 40 halves = 80 bytes row stride (16B aligned)

__global__ __launch_bounds__(256) void gemm_wmma_kernel(
    const float* __restrict__ A, const float* __restrict__ W,
    const float* __restrict__ bias, float* __restrict__ C,
    int M, int N, int K, int act)
{
  __shared__ _Float16 As [BM][LDK];
  __shared__ _Float16 Bts[BN][LDK];

  const int tid  = threadIdx.x;
  const int wave = tid >> 5;
  const int lane = tid & 31;
  const int m0 = blockIdx.y * BM;
  const int n0 = blockIdx.x * BN;
  const int wm = (wave >> 1) * 32;     // wave M offset in block tile
  const int wn = (wave & 1) * 32;      // wave N offset in block tile

  // cooperative-load assignments
  const int arow = tid >> 2;           // 0..63 (and +64)
  const int acol = (tid & 3) * 8;      // 0,8,16,24
  const int bn_  = tid & 63;           // B: column n within tile
  const int bk_  = (tid >> 6) * 8;     // B: k group (0,8,16,24)

  // fragment lane decomposition (ISA 7.12.2)
  const int r  = lane & 15;            // row (A) / col (B) within 16
  const int kh = (lane >> 4) * 8;      // A: K half-select {0,8}
  const int kb = (lane >> 4) * 16;     // unused; B also uses kh {0,8}

  v8f acc[2][2] = {};

  for (int k0 = 0; k0 < K; k0 += BK) {
    // ---- stage A tile (128x32) as f16, vector loads + vector stores ----
#pragma unroll
    for (int rr = 0; rr < 2; ++rr) {
      const int row = arow + rr * 64;
      const float* ap = A + (size_t)(m0 + row) * K + (k0 + acol);
      const float4 f0 = *(const float4*)ap;
      const float4 f1 = *(const float4*)(ap + 4);
      v8h h;
      h[0] = (_Float16)f0.x; h[1] = (_Float16)f0.y;
      h[2] = (_Float16)f0.z; h[3] = (_Float16)f0.w;
      h[4] = (_Float16)f1.x; h[5] = (_Float16)f1.y;
      h[6] = (_Float16)f1.z; h[7] = (_Float16)f1.w;
      *(v8h*)&As[row][acol] = h;
    }
    // ---- stage B tile (32x64) transposed into Bts[n][k] ----
    // thread owns column n and K-group bk_: 8 coalesced dword loads (each
    // instruction covers 32 consecutive n across the wave), then ONE
    // contiguous 16B LDS store.
    {
      const float* bp = W + (size_t)(k0 + bk_) * N + (n0 + bn_);
      float f[8];
#pragma unroll
      for (int i = 0; i < 8; ++i) f[i] = bp[(size_t)i * N];
      v8h h;
#pragma unroll
      for (int i = 0; i < 8; ++i) h[i] = (_Float16)f[i];
      *(v8h*)&Bts[bn_][bk_] = h;
    }
    // prefetch next K tile while this one is consumed
    if (k0 + BK < K) {
      __builtin_prefetch(A + (size_t)(m0 + arow) * K + (k0 + BK + acol), 0, 0);
      __builtin_prefetch(W + (size_t)(k0 + BK + bk_) * N + (n0 + bn_), 0, 0);
    }
    __syncthreads();

    // ---- fragments: each is two 16B ds loads ----
    v16h af[2], bf[2];
#pragma unroll
    for (int j = 0; j < 2; ++j) {
      const _Float16* pa = &As[wm + j * 16 + r][0];
      const v8h alo = *(const v8h*)(pa + kh);        // K = kh..kh+7
      const v8h ahi = *(const v8h*)(pa + 16 + kh);   // K = 16+kh..23+kh
      af[j] = __builtin_shufflevector(alo, ahi,
              0,1,2,3,4,5,6,7,8,9,10,11,12,13,14,15);
      const _Float16* pb = &Bts[wn + j * 16 + r][0];
      const v8h blo = *(const v8h*)(pb + kh);        // K = kh..kh+7
      const v8h bhi = *(const v8h*)(pb + 16 + kh);   // K = 16+kh..23+kh
      bf[j] = __builtin_shufflevector(blo, bhi,
              0,1,2,3,4,5,6,7,8,9,10,11,12,13,14,15);
    }
    // ---- 2x2 wave tile: 4 WMMAs with full fragment reuse ----
#pragma unroll
    for (int j = 0; j < 2; ++j)
#pragma unroll
      for (int i = 0; i < 2; ++i)
        acc[j][i] = __builtin_amdgcn_wmma_f32_16x16x32_f16(
            false, af[j], false, bf[i], (short)0, acc[j][i], false, false);
    __syncthreads();
  }

  // ---- epilogue: C/D layout (VGPR rr -> M = rr + 8*(lane>=16); N = lane&15)
#pragma unroll
  for (int j = 0; j < 2; ++j) {
    const int mb = m0 + wm + j * 16 + (lane >> 4) * 8;
#pragma unroll
    for (int i = 0; i < 2; ++i) {
      const int n = n0 + wn + i * 16 + (lane & 15);
      const float bn = bias ? bias[n] : 0.0f;
#pragma unroll
      for (int rr = 0; rr < 8; ++rr) {
        float v = acc[j][i][rr] + bn;
        if (act == 1) v = 0.5f * v * (1.0f + erff(v * 0.7071067811865476f));
        C[(size_t)(mb + rr) * N + n] = v;
      }
    }
  }
}

// ---------------------------------------------------------------------------
// Attention: one block per (b, h, q-row).  Q/K/V stored [B, S, H*DK].
// ---------------------------------------------------------------------------
__global__ __launch_bounds__(256) void attention_kernel(
    const float* __restrict__ Q, const float* __restrict__ Km,
    const float* __restrict__ Vm, float* __restrict__ O, int causal)
{
  const int qi  = blockIdx.x;          // query position 0..SEQ-1
  const int bh  = blockIdx.y;          // b*NHEAD + h
  const int b   = bh >> 3;
  const int h   = bh & 7;
  const int tid = threadIdx.x;

  __shared__ float qs[DKH];
  __shared__ float sc[SEQ];
  __shared__ float red[256];
  __shared__ float pv[256];

  const size_t qoff = ((size_t)b * SEQ + qi) * DMODEL + h * DKH;
  if (tid < DKH) qs[tid] = Q[qoff + tid];
  __syncthreads();

  float s[2];
  float lmax = -3.0e38f;
#pragma unroll
  for (int j = 0; j < 2; ++j) {
    const int k = tid + j * 256;
    const float* kr = Km + ((size_t)b * SEQ + k) * DMODEL + h * DKH;
    float dot = 0.0f;
#pragma unroll 8
    for (int d = 0; d < DKH; ++d) dot += qs[d] * kr[d];
    float v = dot * 0.125f;              // 1/sqrt(64)
    if (causal && k > qi) v = -1.0e9f;
    s[j] = v;
    lmax = fmaxf(lmax, v);
  }

  red[tid] = lmax;
  __syncthreads();
  for (int st = 128; st > 0; st >>= 1) {
    if (tid < st) red[tid] = fmaxf(red[tid], red[tid + st]);
    __syncthreads();
  }
  const float m = red[0];
  __syncthreads();

  float lsum = 0.0f;
#pragma unroll
  for (int j = 0; j < 2; ++j) {
    const float e = __expf(s[j] - m);
    sc[tid + j * 256] = e;
    lsum += e;
  }
  red[tid] = lsum;
  __syncthreads();
  for (int st = 128; st > 0; st >>= 1) {
    if (tid < st) red[tid] += red[tid + st];
    __syncthreads();
  }
  const float inv = 1.0f / red[0];
  __syncthreads();

  // P @ V: thread (chunk, d) accumulates 128 keys for one of 64 dims
  const int d     = tid & 63;
  const int chunk = tid >> 6;
  float acc = 0.0f;
  const float* vbase = Vm + (size_t)b * SEQ * DMODEL + h * DKH + d;
  const int kbeg = chunk * (SEQ / 4);
  for (int k = kbeg; k < kbeg + SEQ / 4; ++k)
    acc += sc[k] * vbase[(size_t)k * DMODEL];
  pv[tid] = acc;
  __syncthreads();
  if (tid < DKH)
    O[qoff + tid] = (pv[tid] + pv[tid + 64] + pv[tid + 128] + pv[tid + 192]) * inv;
}

// ---------------------------------------------------------------------------
// Fused residual + LayerNorm:  O[t] = g * norm(X[t] + R[t]) + be
// (register-held values, so O may alias X)
// ---------------------------------------------------------------------------
__global__ __launch_bounds__(256) void add_ln_kernel(
    const float* __restrict__ X, const float* __restrict__ R,
    const float* __restrict__ g, const float* __restrict__ be,
    float* __restrict__ O)
{
  const int t   = blockIdx.x;
  const int tid = threadIdx.x;
  __shared__ float red[256];
  const size_t base = (size_t)t * DMODEL;

  const float x0 = X[base + tid]       + R[base + tid];
  const float x1 = X[base + tid + 256] + R[base + tid + 256];

  red[tid] = x0 + x1;
  __syncthreads();
  for (int st = 128; st > 0; st >>= 1) {
    if (tid < st) red[tid] += red[tid + st];
    __syncthreads();
  }
  const float mean = red[0] * (1.0f / DMODEL);
  __syncthreads();

  const float d0 = x0 - mean, d1 = x1 - mean;
  red[tid] = d0 * d0 + d1 * d1;
  __syncthreads();
  for (int st = 128; st > 0; st >>= 1) {
    if (tid < st) red[tid] += red[tid + st];
    __syncthreads();
  }
  const float rstd = rsqrtf(red[0] * (1.0f / DMODEL) + 1e-5f);

  O[base + tid]       = g[tid]       * d0 * rstd + be[tid];
  O[base + tid + 256] = g[tid + 256] * d1 * rstd + be[tid + 256];
}

// ---------------------------------------------------------------------------
// Host-side orchestration
// ---------------------------------------------------------------------------
extern "C" void kernel_launch(void* const* d_in, const int* in_sizes, int n_in,
                              void* d_out, int out_size, void* d_ws, size_t ws_size,
                              hipStream_t stream) {
  (void)in_sizes; (void)n_in; (void)out_size; (void)ws_size;

  // ---- inputs (setup_inputs dict order, nested dicts flattened in order) ----
  const int*   src      = (const int*)d_in[0];
  const int*   tgt      = (const int*)d_in[1];
  // d_in[2] src_mask (all ones) and d_in[3] tgt_mask (causal) handled implicitly
  const float* emb_src  = (const float*)d_in[4];
  const float* emb_tgt  = (const float*)d_in[5];
  const float* pe       = (const float*)d_in[6];
  // enc_params: 7..22  | dec_params: 23..48
  const float* fc_w     = (const float*)d_in[49];
  const float* fc_b     = (const float*)d_in[50];

  auto P = [&](int i) { return (const float*)d_in[i]; };
  const size_t DD = (size_t)DMODEL * DMODEL;   // 262144
  const size_t DF = (size_t)DMODEL * FDIM;     // 1048576

  // ---- workspace layout (floats) ----
  float* ws = (float*)d_ws;
  const size_t TD = (size_t)NTOK * DMODEL;     // 1M floats
  float* xe = ws;            // encoder stream (ends as encoder output)
  float* yd = ws + 1 * TD;   // decoder stream
  float* qb = ws + 2 * TD;
  float* kb = ws + 3 * TD;
  float* vb = ws + 4 * TD;
  float* ab = ws + 5 * TD;   // attention output
  float* tb = ws + 6 * TD;   // projection / ffn-out temp
  float* fb = ws + 7 * TD;   // ffn hidden: NTOK x FDIM = 4M floats

  auto gemm = [&](const float* A, const float* W, const float* bias, float* C,
                  int M, int N, int K, int act) {
    gemm_wmma_kernel<<<dim3(N / BN, M / BM), 256, 0, stream>>>(A, W, bias, C, M, N, K, act);
  };
  const dim3 attn_grid(SEQ, BSZ * NHEAD);

  // ======================= Encoder =======================
  embed_kernel<<<NTOK, 256, 0, stream>>>(src, emb_src, pe, xe);
  for (int l = 0; l < NLAYER; ++l) {
    const float* wq = P(7)  + l * DD;  const float* bq  = P(8)  + l * DMODEL;
    const float* wk = P(9)  + l * DD;  const float* bk  = P(10) + l * DMODEL;
    const float* wv = P(11) + l * DD;  const float* bv  = P(12) + l * DMODEL;
    const float* wo = P(13) + l * DD;  const float* bo  = P(14) + l * DMODEL;
    const float* w1 = P(15) + l * DF;  const float* b1  = P(16) + l * FDIM;
    const float* w2 = P(17) + l * DF;  const float* b2  = P(18) + l * DMODEL;
    const float* g1 = P(19) + l * DMODEL; const float* be1 = P(20) + l * DMODEL;
    const float* g2 = P(21) + l * DMODEL; const float* be2 = P(22) + l * DMODEL;

    gemm(xe, wq, bq, qb, NTOK, DMODEL, DMODEL, 0);
    gemm(xe, wk, bk, kb, NTOK, DMODEL, DMODEL, 0);
    gemm(xe, wv, bv, vb, NTOK, DMODEL, DMODEL, 0);
    attention_kernel<<<attn_grid, 256, 0, stream>>>(qb, kb, vb, ab, 0);
    gemm(ab, wo, bo, tb, NTOK, DMODEL, DMODEL, 0);
    add_ln_kernel<<<NTOK, 256, 0, stream>>>(xe, tb, g1, be1, xe);
    gemm(xe, w1, b1, fb, NTOK, FDIM, DMODEL, 1);   // GELU
    gemm(fb, w2, b2, tb, NTOK, DMODEL, FDIM, 0);
    add_ln_kernel<<<NTOK, 256, 0, stream>>>(xe, tb, g2, be2, xe);
  }

  // ======================= Decoder =======================
  embed_kernel<<<NTOK, 256, 0, stream>>>(tgt, emb_tgt, pe, yd);
  for (int l = 0; l < NLAYER; ++l) {
    const float* sq = P(23) + l * DD;  const float* sbq = P(24) + l * DMODEL;
    const float* sk = P(25) + l * DD;  const float* sbk = P(26) + l * DMODEL;
    const float* sv = P(27) + l * DD;  const float* sbv = P(28) + l * DMODEL;
    const float* so = P(29) + l * DD;  const float* sbo = P(30) + l * DMODEL;
    const float* cq = P(31) + l * DD;  const float* cbq = P(32) + l * DMODEL;
    const float* ck = P(33) + l * DD;  const float* cbk = P(34) + l * DMODEL;
    const float* cv = P(35) + l * DD;  const float* cbv = P(36) + l * DMODEL;
    const float* co = P(37) + l * DD;  const float* cbo = P(38) + l * DMODEL;
    const float* w1 = P(39) + l * DF;  const float* b1  = P(40) + l * FDIM;
    const float* w2 = P(41) + l * DF;  const float* b2  = P(42) + l * DMODEL;
    const float* g1 = P(43) + l * DMODEL; const float* be1 = P(44) + l * DMODEL;
    const float* g2 = P(45) + l * DMODEL; const float* be2 = P(46) + l * DMODEL;
    const float* g3 = P(47) + l * DMODEL; const float* be3 = P(48) + l * DMODEL;

    // masked self-attention
    gemm(yd, sq, sbq, qb, NTOK, DMODEL, DMODEL, 0);
    gemm(yd, sk, sbk, kb, NTOK, DMODEL, DMODEL, 0);
    gemm(yd, sv, sbv, vb, NTOK, DMODEL, DMODEL, 0);
    attention_kernel<<<attn_grid, 256, 0, stream>>>(qb, kb, vb, ab, 1);
    gemm(ab, so, sbo, tb, NTOK, DMODEL, DMODEL, 0);
    add_ln_kernel<<<NTOK, 256, 0, stream>>>(yd, tb, g1, be1, yd);

    // cross-attention (K/V from encoder output)
    gemm(yd, cq, cbq, qb, NTOK, DMODEL, DMODEL, 0);
    gemm(xe, ck, cbk, kb, NTOK, DMODEL, DMODEL, 0);
    gemm(xe, cv, cbv, vb, NTOK, DMODEL, DMODEL, 0);
    attention_kernel<<<attn_grid, 256, 0, stream>>>(qb, kb, vb, ab, 0);
    gemm(ab, co, cbo, tb, NTOK, DMODEL, DMODEL, 0);
    add_ln_kernel<<<NTOK, 256, 0, stream>>>(yd, tb, g2, be2, yd);

    // FFN
    gemm(yd, w1, b1, fb, NTOK, FDIM, DMODEL, 1);   // GELU
    gemm(fb, w2, b2, tb, NTOK, DMODEL, FDIM, 0);
    add_ln_kernel<<<NTOK, 256, 0, stream>>>(yd, tb, g3, be3, yd);
  }

  // ======================= Logits =======================
  gemm(yd, fc_w, fc_b, (float*)d_out, NTOK, VOCAB, DMODEL, 0);
}
